// GATInductiveNet_566935683604
// MI455X (gfx1250) — matrix operand order repaired and checked by reading, not compiled, not run on hardware
//
#include <hip/hip_runtime.h>
#include <hip/hip_bf16.h>
#include <math.h>

typedef _Float16 v16h __attribute__((ext_vector_type(16)));
typedef _Float16 v8h  __attribute__((ext_vector_type(8)));
typedef float    v8f  __attribute__((ext_vector_type(8)));
typedef unsigned int u32x4 __attribute__((ext_vector_type(4)));
typedef int      i32x4 __attribute__((ext_vector_type(4)));
typedef int      i32x8 __attribute__((ext_vector_type(8)));

#define NND 4096
#define NEGV (-1.0e9f)

#if __has_builtin(__builtin_amdgcn_tensor_load_to_lds) && __has_builtin(__builtin_amdgcn_s_wait_tensorcnt)
#define HAVE_TDM 1
#else
#define HAVE_TDM 0
#endif

// ---------------------------------------------------------------------------
// TDM: 2D tile load  hT[rows][4096] f16 -> LDS [rows][32] f16 (contiguous)
// D# fields per CDNA5 ISA ch.8: group0 {count,lds_addr,global_addr,type=2},
// group1 {data_size=2B, tensor_dim0=4096, tensor_dim1=rows, tile=32 x rows,
//         tensor_dim0_stride=4096}. 6-arg builtin: trailing i32x8 group unused
// for 2D tiles (zeros), cpol=0.
// ---------------------------------------------------------------------------
#if HAVE_TDM
__device__ __forceinline__ void tdm_load_tile(const _Float16* g, unsigned ldsOff, int rows) {
    unsigned long long ga = (unsigned long long)g;
    u32x4 g0 = { 1u,                                  // count=1, user mode
                 ldsOff,                              // lds_addr (bytes)
                 (unsigned)(ga & 0xffffffffu),        // global_addr[31:0]
                 (unsigned)((ga >> 32) & 0x1ffffffu) | (2u << 30) }; // addr[56:32] | type=2
    i32x8 g1 = { (int)(1u << 16),        // data_size=1 (2 bytes) at bits[17:16]
                 (int)(4096u << 16),     // tensor_dim0 low16 at bits[63:48]
                 rows << 16,             // tensor_dim1 low16 at bits[95:80]
                 32 << 16,               // tile_dim0 = 32 at bits[127:112]
                 rows,                   // tile_dim1 at bits[143:128]
                 4096,                   // tensor_dim0_stride low32
                 0, 0 };
    i32x4 z4 = {0, 0, 0, 0};
    i32x8 z8 = {0, 0, 0, 0, 0, 0, 0, 0};
    __builtin_amdgcn_tensor_load_to_lds(g0, g1, z4, z4, z8, 0);
}
#endif

// ---------------------------------------------------------------------------
// K0: bit-pack adjacency: adj[4096][4096] i32 -> bits[4096][128] u32
// ---------------------------------------------------------------------------
__global__ void k_pack_adj(const int* __restrict__ adj, unsigned* __restrict__ bits) {
    int w = blockIdx.x * 256 + threadIdx.x;     // 4096*128 words
    int n = w >> 7, wi = w & 127;
    const int* row = adj + (size_t)n * NND + wi * 32;
    unsigned b = 0;
    #pragma unroll
    for (int j = 0; j < 32; ++j) b |= (row[j] != 0 ? 1u : 0u) << j;
    bits[w] = b;
}

// ---------------------------------------------------------------------------
// K1: x[4096][50] f32 -> Xh[4096][64] f16 (zero pad K)
// ---------------------------------------------------------------------------
__global__ void k_conv_x(const float* __restrict__ x, _Float16* __restrict__ Xh) {
    int i = blockIdx.x * 256 + threadIdx.x;     // 4096*64
    int k = i & 63, n = i >> 6;
    Xh[i] = (_Float16)(k < 50 ? x[(size_t)n * 50 + k] : 0.f);
}

// ---------------------------------------------------------------------------
// K2: W[H][Kin][Nin] f32 -> WT[H][Npad][Kpad] f16  (transposed + padded)
// ---------------------------------------------------------------------------
__global__ void k_conv_wt(const float* __restrict__ W, _Float16* __restrict__ WT,
                          int Kin, int Kpad, int Nin, int Npad) {
    size_t i = (size_t)blockIdx.x * 256 + threadIdx.x;   // H*Npad*Kpad
    int k  = (int)(i % Kpad);
    int o  = (int)((i / Kpad) % Npad);
    int hd = (int)(i / ((size_t)Kpad * Npad));
    float v = (k < Kin && o < Nin) ? W[((size_t)hd * Kin + k) * Nin + o] : 0.f;
    WT[i] = (_Float16)v;
}

// ---------------------------------------------------------------------------
// K3: h = A x W via WMMA, stored transposed: hT[H][NT*16][4096] f16
//   A: [4096][K] f16 row-major; WT: [H][NT*16][K] f16 (o-major, k contiguous)
//   grid (32, H), block 256: wave w does node tile n0=(bx*8+w)*16, all o tiles
// ---------------------------------------------------------------------------
template <int NT>
__global__ __launch_bounds__(256) void k_gemm_hT(const _Float16* __restrict__ A, int K,
                                                 const _Float16* __restrict__ WT,
                                                 _Float16* __restrict__ hT) {
    const int lane = threadIdx.x & 31, wave = threadIdx.x >> 5;
    const int hi = lane >> 4, q = lane & 15;
    const int hd = blockIdx.y;
    const int n0 = (blockIdx.x * 8 + wave) * 16;
    const _Float16* arow = A + (size_t)(n0 + q) * K;

    v8f acc[NT];
    #pragma unroll
    for (int ot = 0; ot < NT; ++ot)
        #pragma unroll
        for (int r = 0; r < 8; ++r) acc[ot][r] = 0.f;

    for (int k0 = 0; k0 < K; k0 += 32) {
        // 16-bit A fragment: lanes0-15 rows, K {0..15}; lanes16-31 {16..31}
        union { v16h v; v8h h[2]; } a;
        a.v = *(const v16h*)(arow + k0 + hi * 16);
        #pragma unroll
        for (int ot = 0; ot < NT; ++ot) {
            const _Float16* brow = WT + ((size_t)(hd * NT + ot) * 16 + q) * K + k0 + hi * 16;
            v16h b = *(const v16h*)brow;
            acc[ot] = __builtin_amdgcn_wmma_f32_16x16x32_f16(false, a.v, false, b,
                                                             (short)0, acc[ot], false, false);
        }
    }
    #pragma unroll
    for (int ot = 0; ot < NT; ++ot) {
        v8h o16;
        #pragma unroll
        for (int r = 0; r < 8; ++r) o16[r] = (_Float16)acc[ot][r];
        _Float16* dst = hT + ((size_t)(hd * NT + ot) * 16 + q) * NND + n0 + hi * 8;
        *(v8h*)dst = o16;
    }
}

// ---------------------------------------------------------------------------
// K4: es/ed[H][4096] = hT . a_src / a_dst  (coalesced along n)
// ---------------------------------------------------------------------------
__global__ void k_esed(const _Float16* __restrict__ hT, const float* __restrict__ as,
                       const float* __restrict__ ad, float* __restrict__ es,
                       float* __restrict__ ed, int Fpad, int Alen) {
    int hd = blockIdx.y;
    int n = blockIdx.x * 256 + threadIdx.x;
    const _Float16* hb = hT + (size_t)hd * Fpad * NND;
    float s = 0.f, d = 0.f;
    for (int o = 0; o < Alen; ++o) {
        float hv = (float)hb[(size_t)o * NND + n];
        s += hv * as[hd * Alen + o];
        d += hv * ad[hd * Alen + o];
    }
    es[hd * NND + n] = s;
    ed[hd * NND + n] = d;
}

// ---------------------------------------------------------------------------
// K5: flash-style masked-softmax aggregation.
//   block = (head hd, 16-node tile n0); 8 waves split FT*16 output cols each.
//   alpha 16x32 tile built in registers -> A frag; B frag from TDM LDS tile.
// ---------------------------------------------------------------------------
template <int FT, int FROWS>
__global__ __launch_bounds__(256) void k_flash_agg(const _Float16* __restrict__ hT,
                                                   const float* __restrict__ es,
                                                   const float* __restrict__ ed,
                                                   const unsigned* __restrict__ adjb,
                                                   float* __restrict__ out, int ostride,
                                                   int colPerHead, size_t headPtrOff) {
    __shared__ __align__(32) _Float16 tile[2][FROWS * 32];
    const int lane = threadIdx.x & 31, wave = threadIdx.x >> 5;
    const int hi = lane >> 4, q = lane & 15;
    const int hd = blockIdx.y;
    const int n0 = blockIdx.x * 16;
    const int n = n0 + q;
    const int kbase = hi * 8;
    const _Float16* hTh = hT + (size_t)hd * FROWS * NND;
    const float* edh = ed + (size_t)hd * NND;
    const float es_n = es[(size_t)hd * NND + n];
    const int obase = wave * (FT * 16);

    v8f acc[FT];
    #pragma unroll
    for (int f = 0; f < FT; ++f)
        #pragma unroll
        for (int r = 0; r < 8; ++r) acc[f][r] = 0.f;
    float row_m = -INFINITY, row_l = 0.f;

    const int T = NND / 32;
#if HAVE_TDM
    if (wave == 0) tdm_load_tile(hTh, 0u, FROWS);
#endif
    for (int t = 0; t < T; ++t) {
        const int m0 = t * 32;
#if HAVE_TDM
        if (wave == 0) {
            if (t + 1 < T) {
                tdm_load_tile(hTh + (t + 1) * 32,
                              (unsigned)(((t + 1) & 1) * FROWS * 32 * 2), FROWS);
                __builtin_amdgcn_s_wait_tensorcnt(1);
            } else {
                __builtin_amdgcn_s_wait_tensorcnt(0);
            }
        }
        __syncthreads();
        const _Float16* tb = &tile[t & 1][0];
#else
        __syncthreads();
        for (int i = threadIdx.x; i < FROWS * 4; i += 256) {
            int o = i >> 2, part = i & 3;
            *(v8h*)&tile[0][o * 32 + part * 8] =
                *(const v8h*)(hTh + (size_t)o * NND + m0 + part * 8);
        }
        __syncthreads();
        const _Float16* tb = &tile[0][0];
#endif
        // --- scores for this lane's row (16 m's, matching A-frag K layout) ---
        unsigned bits = adjb[(size_t)n * 128 + t];
        v8f ev0 = *(const v8f*)(edh + m0 + kbase);
        v8f ev1 = *(const v8f*)(edh + m0 + kbase + 16);
        float e[16];
        #pragma unroll
        for (int j = 0; j < 8; ++j) {
            float t0 = es_n + ev0[j];
            t0 = t0 > 0.f ? t0 : 0.2f * t0;
            e[j] = ((bits >> (kbase + j)) & 1u) ? t0 : NEGV;
            float t1 = es_n + ev1[j];
            t1 = t1 > 0.f ? t1 : 0.2f * t1;
            e[8 + j] = ((bits >> (kbase + 16 + j)) & 1u) ? t1 : NEGV;
        }
        float tmax = e[0];
        #pragma unroll
        for (int j = 1; j < 16; ++j) tmax = fmaxf(tmax, e[j]);
        tmax = fmaxf(tmax, __shfl_xor(tmax, 16));
        float newm = fmaxf(row_m, tmax);
        float scale = expf(row_m - newm);
        float p[16], tsum = 0.f;
        #pragma unroll
        for (int j = 0; j < 16; ++j) { p[j] = expf(e[j] - newm); tsum += p[j]; }
        tsum += __shfl_xor(tsum, 16);
        row_l = row_l * scale + tsum;
        row_m = newm;
        // --- rescale accumulators (C rows r+8*hi live in lane r+8*hi's state) ---
        #pragma unroll
        for (int r = 0; r < 8; ++r) {
            float sr = __shfl(scale, (hi << 3) + r);
            #pragma unroll
            for (int f = 0; f < FT; ++f) acc[f][r] *= sr;
        }
        // --- alpha A-fragment + WMMA against LDS hT tile ---
        v16h af;
        #pragma unroll
        for (int j = 0; j < 16; ++j) af[j] = (_Float16)p[j];
        #pragma unroll
        for (int f = 0; f < FT; ++f) {
            int o = obase + f * 16 + q;
            union { v16h v; v8h h[2]; } bu;
            bu.h[0] = *(const v8h*)(tb + o * 32 + hi * 8);
            bu.h[1] = *(const v8h*)(tb + o * 32 + hi * 8 + 16);
            v16h b = bu.v;
            acc[f] = __builtin_amdgcn_wmma_f32_16x16x32_f16(false, af, false, b,
                                                            (short)0, acc[f], false, false);
        }
        __syncthreads();
    }
    // --- normalize by row sums and write out ---
    float invl = 1.f / row_l;
    #pragma unroll
    for (int r = 0; r < 8; ++r) {
        float ir = __shfl(invl, (hi << 3) + r);
        int nout = n0 + (hi << 3) + r;
        #pragma unroll
        for (int f = 0; f < FT; ++f) {
            int col = hd * colPerHead + obase + f * 16 + q;
            out[headPtrOff * hd + (size_t)nout * ostride + col] = acc[f][r] * ir;
        }
    }
}

// ---------------------------------------------------------------------------
// K6: ELU (+optional residual), write f32 and f16 copies
// ---------------------------------------------------------------------------
__global__ void k_elu_cvt(const float* agg, const float* resid, float* xf,
                          _Float16* xh, int hasResid) {
    size_t i = (size_t)blockIdx.x * 256 + threadIdx.x;
    float v = agg[i];
    v = v > 0.f ? v : expm1f(v);
    if (hasResid) v += resid[i];
    xf[i] = v;
    xh[i] = (_Float16)v;
}

// ---------------------------------------------------------------------------
// K7: mean over 6 heads + log_softmax over 121 classes
// ---------------------------------------------------------------------------
__global__ void k_final(const float* __restrict__ agg3, float* __restrict__ out) {
    int n = blockIdx.x, o = threadIdx.x;   // 128 threads
    __shared__ float red[128];
    float val = 0.f, v = -INFINITY;
    if (o < 121) {
        float s = 0.f;
        #pragma unroll
        for (int hd = 0; hd < 6; ++hd) s += agg3[((size_t)hd * NND + n) * 128 + o];
        val = s * (1.f / 6.f);
        v = val;
    }
    red[o] = v; __syncthreads();
    for (int st = 64; st > 0; st >>= 1) { if (o < st) red[o] = fmaxf(red[o], red[o + st]); __syncthreads(); }
    float mx = red[0]; __syncthreads();
    red[o] = (o < 121) ? expf(val - mx) : 0.f; __syncthreads();
    for (int st = 64; st > 0; st >>= 1) { if (o < st) red[o] += red[o + st]; __syncthreads(); }
    float ls = logf(red[0]);
    if (o < 121) out[(size_t)n * 121 + o] = val - mx - ls;
}

// ---------------------------------------------------------------------------
extern "C" void kernel_launch(void* const* d_in, const int* in_sizes, int n_in,
                              void* d_out, int out_size, void* d_ws, size_t ws_size,
                              hipStream_t stream) {
    (void)in_sizes; (void)n_in; (void)out_size; (void)ws_size;
    const float* x   = (const float*)d_in[0];
    const int*   adj = (const int*)d_in[1];
    const float* W1  = (const float*)d_in[2];
    const float* a1s = (const float*)d_in[3];
    const float* a1d = (const float*)d_in[4];
    const float* W2  = (const float*)d_in[5];
    const float* a2s = (const float*)d_in[6];
    const float* a2d = (const float*)d_in[7];
    const float* W3  = (const float*)d_in[8];
    const float* a3s = (const float*)d_in[9];
    const float* a3d = (const float*)d_in[10];
    float* out = (float*)d_out;

    char* ws = (char*)d_ws;
    unsigned*  adjb = (unsigned*)(ws + 0);                 //  2 MB
    _Float16*  Xh   = (_Float16*)(ws + 2097152);           //  512 KB
    _Float16*  WT   = (_Float16*)(ws + 2621440);           //  2 MB (reused per layer)
    _Float16*  hT   = (_Float16*)(ws + 4718592);           //  8 MB (reused per layer)
    float*     es   = (float*)   (ws + 13107200);          //  96 KB
    float*     ed   = (float*)   (ws + 13205504);          //  96 KB
    float*     agg  = (float*)   (ws + 13303808);          // 16 MB
    float*     x1f  = (float*)   (ws + 30081024);          // 16 MB
    _Float16*  xhc  = (_Float16*)(ws + 46858240);          //  8 MB
    float*     agg3 = (float*)   (ws + 55246848);          // 12 MB

    k_pack_adj<<<2048, 256, 0, stream>>>(adj, adjb);
    k_conv_x<<<1024, 256, 0, stream>>>(x, Xh);

    // ---- Layer 1: H=4, K=64(pad of 50), F=256 ----
    k_conv_wt<<<256, 256, 0, stream>>>(W1, WT, 50, 64, 256, 256);
    k_gemm_hT<16><<<dim3(32, 4), 256, 0, stream>>>(Xh, 64, WT, hT);
    k_esed<<<dim3(16, 4), 256, 0, stream>>>(hT, a1s, a1d, es, ed, 256, 256);
    k_flash_agg<2, 256><<<dim3(256, 4), 256, 0, stream>>>(hT, es, ed, adjb, agg,
                                                          1024, 256, (size_t)0);
    k_elu_cvt<<<16384, 256, 0, stream>>>(agg, nullptr, x1f, xhc, 0);

    // ---- Layer 2: H=4, K=1024, F=256, residual ----
    k_conv_wt<<<4096, 256, 0, stream>>>(W2, WT, 1024, 1024, 256, 256);
    k_gemm_hT<16><<<dim3(32, 4), 256, 0, stream>>>(xhc, 1024, WT, hT);
    k_esed<<<dim3(16, 4), 256, 0, stream>>>(hT, a2s, a2d, es, ed, 256, 256);
    k_flash_agg<2, 256><<<dim3(256, 4), 256, 0, stream>>>(hT, es, ed, adjb, agg,
                                                          1024, 256, (size_t)0);
    k_elu_cvt<<<16384, 256, 0, stream>>>(agg, x1f, x1f, xhc, 1);

    // ---- Layer 3: H=6, K=1024, F=128(pad of 121), head-mean + log_softmax ----
    k_conv_wt<<<3072, 256, 0, stream>>>(W3, WT, 1024, 1024, 121, 128);
    k_gemm_hT<8><<<dim3(32, 6), 256, 0, stream>>>(xhc, 1024, WT, hT);
    k_esed<<<dim3(16, 6), 256, 0, stream>>>(hT, a3s, a3d, es, ed, 128, 121);
    k_flash_agg<1, 128><<<dim3(256, 6), 256, 0, stream>>>(hT, es, ed, adjb, agg3,
                                                          128, 0, (size_t)NND * 128);
    k_final<<<4096, 128, 0, stream>>>(agg3, out);
}
